// SparseKANConv3D_19885698581240
// MI455X (gfx1250) — matrix operand order repaired and checked by reading, not compiled, not run on hardware
//
#include <hip/hip_runtime.h>
#include <cstdint>

// ---------------- problem constants (from reference) ----------------
#define NUM_K   27
#define C_IN    32
#define C_OUT   64
#define NVOX    200000
#define NP      100000
#define RDIM    256                    // C_IN * (GRID_SIZE + SPLINE_ORDER) = 32*8
#define TM      128                    // pairs per block (8 waves x 16 rows)
#define TILES   ((NP + TM - 1) / TM)   // 782

// padded LDS strides (bank-conflict free for 16-lane strided b128 loads,
// while keeping 16B alignment for uint4/float4 accesses)
#define SW_STRIDE (RDIM + 8)           // 264 shorts -> 528B row stride
#define SB_STRIDE (C_IN + 8)           // 40 shorts  -> 80B row stride
#define SX_STRIDE (C_IN + 4)           // 36 floats  -> 144B row stride

#if defined(__has_builtin)
#if __has_builtin(__builtin_amdgcn_global_load_async_to_lds_b128)
#define HAVE_ASYNC_GATHER 1
#endif
#if __has_builtin(__builtin_amdgcn_s_wait_asynccnt)
#define HAVE_WAIT_ASYNC_BUILTIN 1
#endif
#endif

typedef __attribute__((ext_vector_type(16))) __bf16 v16bf;
typedef __attribute__((ext_vector_type(8)))  float  v8f;
typedef int v4i __attribute__((vector_size(16)));

#if HAVE_ASYNC_GATHER
typedef __attribute__((address_space(1))) v4i* as1_v4i_ptr;
typedef __attribute__((address_space(3))) v4i* as3_v4i_ptr;
#endif

union BF16Vec { v16bf v; unsigned short s[16]; uint4 q[2]; };

__device__ __forceinline__ unsigned short f2bf(float f) {
    unsigned int u = __float_as_uint(f);
    u += 0x7FFFu + ((u >> 16) & 1u);   // round-to-nearest-even truncation
    return (unsigned short)(u >> 16);
}

// Cox-de Boor, order 3, uniform grid h=0.4 on [-1,1] with 3 pad knots each side.
// g_i = (i-3)*h - 1, i = 0..11. All divisors fold to compile-time reciprocals.
__device__ __forceinline__ void bspline8(float x, float* o) {
    const float h = 0.4f;
    float b[11];
#pragma unroll
    for (int i = 0; i < 11; ++i) {
        float gi  = (float)(i - 3) * h - 1.0f;
        float gi1 = (float)(i - 2) * h - 1.0f;
        b[i] = (x >= gi && x < gi1) ? 1.0f : 0.0f;
    }
#pragma unroll
    for (int k = 1; k <= 3; ++k) {
        float inv = 1.0f / ((float)k * h);
#pragma unroll
        for (int i = 0; i < 11 - k; ++i) {
            float gi   = (float)(i - 3) * h - 1.0f;       // g[i]
            float gik1 = (float)(i + k - 2) * h - 1.0f;   // g[i+k+1]
            b[i] = ((x - gi) * b[i] + (gik1 - x) * b[i + 1]) * inv;
        }
    }
#pragma unroll
    for (int i = 0; i < 8; ++i) o[i] = b[i];
}

__global__ void zero_f4(float4* __restrict__ p, int n4) {
    int i = blockIdx.x * blockDim.x + threadIdx.x;
    if (i < n4) p[i] = float4{0.f, 0.f, 0.f, 0.f};
}

__global__ __launch_bounds__(256) void kan_subm_conv3d(
    const float* __restrict__ features,
    const float* __restrict__ base_w,     // [NUM_K][C_OUT][C_IN]
    const float* __restrict__ spline_w,   // [NUM_K][C_OUT][RDIM]
    const int*   __restrict__ inp_idx,    // [NUM_K][NP]
    const int*   __restrict__ out_idx,    // [NUM_K][NP]
    float*       __restrict__ out)        // [NVOX][C_OUT]
{
    __shared__ unsigned short sW[C_OUT * SW_STRIDE];  // spline weights, bf16
    __shared__ unsigned short sB[C_OUT * SB_STRIDE];  // base weights, bf16
    __shared__ float          sX[TM * SX_STRIDE];     // gathered features

    const int k    = blockIdx.y;
    const int tile = blockIdx.x;
    const int tid  = threadIdx.x;

    const int* iidx = inp_idx + (size_t)k * NP;
    __builtin_prefetch(out_idx + (size_t)k * NP + tile * TM, 0, 0);

#if HAVE_ASYNC_GATHER
    // ---- phase 0: fire the feature gather via the async DMA path, so it
    // overlaps the (much larger) weight staging below. 128 rows x 128B,
    // each thread issues 4 b128 async-to-LDS ops (ASYNCcnt-tracked).
#pragma unroll
    for (int it = 0; it < 4; ++it) {
        int i  = it * 256 + tid;              // [0, 1024)
        int r  = i >> 3, c4 = i & 7;
        int m  = tile * TM + r;
        if (m >= NP) m = NP - 1;              // clamped dups are never scattered
        const float* g = features + (size_t)iidx[m] * C_IN + c4 * 4;
        float*       l = &sX[r * SX_STRIDE + c4 * 4];
        __builtin_amdgcn_global_load_async_to_lds_b128(
            (as1_v4i_ptr)(const void*)g, (as3_v4i_ptr)(void*)l, 0, 0);
    }
#endif

    // ---- phase 1: bf16 weight staging ----
    const float* wsp = spline_w + (size_t)k * (C_OUT * RDIM);
    for (int i = tid; i < C_OUT * RDIM; i += 256) {
        int n = i >> 8, r = i & 255;
        sW[n * SW_STRIDE + r] = f2bf(wsp[i]);
    }
    const float* wbs = base_w + (size_t)k * (C_OUT * C_IN);
    for (int i = tid; i < C_OUT * C_IN; i += 256) {
        int n = i >> 5, c = i & 31;
        sB[n * SB_STRIDE + c] = f2bf(wbs[i]);
    }

#if HAVE_ASYNC_GATHER
#if HAVE_WAIT_ASYNC_BUILTIN
    __builtin_amdgcn_s_wait_asynccnt(0);
#else
    asm volatile("s_wait_asynccnt 0x0" ::: "memory");
#endif
#else
    // ---- fallback: synchronous coalesced gather (128B per row) ----
    for (int i = tid; i < TM * (C_IN / 4); i += 256) {
        int r = i >> 3, c4 = i & 7;
        int m = tile * TM + r;
        if (m >= NP) m = NP - 1;
        float4 v = reinterpret_cast<const float4*>(
            features + (size_t)iidx[m] * C_IN)[c4];
        reinterpret_cast<float4*>(&sX[r * SX_STRIDE])[c4] = v;
    }
#endif
    __syncthreads();

    const int lane = tid & 31;
    const int wid  = tid >> 5;
    const int hi   = lane >> 4;      // lane half (K-half selector for A/B)
    const int ln   = lane & 15;      // M index (A) / N index (B,C)
    const int lr   = wid * 16 + ln;  // this lane's local pair row

    v8f acc[4];
#pragma unroll
    for (int nt = 0; nt < 4; ++nt)
        acc[nt] = v8f{0.f, 0.f, 0.f, 0.f, 0.f, 0.f, 0.f, 0.f};

    // ---- SiLU base contribution: D += silu(X)[16x32] * Wb^T[32x64] ----
    {
        BF16Vec a;
        const int cb = hi * 8;       // A layout: elems 0..7 -> K=cb..cb+7, 8..15 -> K=cb+16..cb+23
#pragma unroll
        for (int t = 0; t < 8; ++t) {
            float x0 = sX[lr * SX_STRIDE + cb + t];
            float x1 = sX[lr * SX_STRIDE + cb + 16 + t];
            a.s[t]     = f2bf(x0 * __builtin_amdgcn_rcpf(1.0f + __expf(-x0)));
            a.s[8 + t] = f2bf(x1 * __builtin_amdgcn_rcpf(1.0f + __expf(-x1)));
        }
        const int kb = hi * 8;       // B layout: elems 0..7 -> K=kb..kb+7, 8..15 -> K=kb+16..kb+23
#pragma unroll
        for (int nt = 0; nt < 4; ++nt) {
            BF16Vec b;
            const uint4* p = reinterpret_cast<const uint4*>(
                &sB[(nt * 16 + ln) * SB_STRIDE + kb]);
            b.q[0] = p[0]; b.q[1] = p[2];
            acc[nt] = __builtin_amdgcn_wmma_f32_16x16x32_bf16(
                false, a.v, false, b.v, (short)0, acc[nt], false, false);
        }
    }

    // ---- spline contribution: D += bases[16x256] * Ws^T[256x64], 8 K-chunks.
    // Unroll x2: two independent A register sets let the next chunk's bspline
    // VALU fill the WMMA->VALU WAR-hazard shadow instead of v_nops.
#pragma unroll 2
    for (int kc = 0; kc < 8; ++kc) {
        BF16Vec a;
        // K-chunk kc covers r = kc*32 .. kc*32+31 -> channels 4kc..4kc+3, 8 bases each.
        // This lane needs channels c1 = 4kc+hi (elems 0..7) and c1+2 (elems 8..15).
        const int c1 = 4 * kc + hi;
        float f8[8];
        bspline8(sX[lr * SX_STRIDE + c1], f8);
#pragma unroll
        for (int t = 0; t < 8; ++t) a.s[t] = f2bf(f8[t]);
        bspline8(sX[lr * SX_STRIDE + c1 + 2], f8);
#pragma unroll
        for (int t = 0; t < 8; ++t) a.s[8 + t] = f2bf(f8[t]);

        const int r0 = kc * 32 + hi * 8;
#pragma unroll
        for (int nt = 0; nt < 4; ++nt) {
            BF16Vec b;
            const uint4* p = reinterpret_cast<const uint4*>(
                &sW[(nt * 16 + ln) * SW_STRIDE + r0]);
            b.q[0] = p[0]; b.q[1] = p[2];
            acc[nt] = __builtin_amdgcn_wmma_f32_16x16x32_bf16(
                false, a.v, false, b.v, (short)0, acc[nt], false, false);
        }
    }

    // ---- scatter-add: C/D layout VGPR j -> row M = j + hi*8, col N = ln ----
    const int m_base = tile * TM + wid * 16;
#pragma unroll
    for (int j = 0; j < 8; ++j) {
        int m = m_base + hi * 8 + j;
        if (m < NP) {
            int orow = out_idx[(size_t)k * NP + m] * C_OUT;
#pragma unroll
            for (int nt = 0; nt < 4; ++nt) {
                __hip_atomic_fetch_add(&out[orow + nt * 16 + ln], acc[nt][j],
                                       __ATOMIC_RELAXED, __HIP_MEMORY_SCOPE_AGENT);
            }
        }
    }
}

extern "C" void kernel_launch(void* const* d_in, const int* in_sizes, int n_in,
                              void* d_out, int out_size, void* d_ws, size_t ws_size,
                              hipStream_t stream) {
    const float* features = (const float*)d_in[0];
    // d_in[1] is `grid` — uniform, reproduced in closed form inside bspline8
    const float* base_w   = (const float*)d_in[2];
    const float* spline_w = (const float*)d_in[3];
    const int*   inp_idx  = (const int*)d_in[4];
    const int*   out_idx  = (const int*)d_in[5];
    float* out = (float*)d_out;

    const int n4 = (NVOX * C_OUT) / 4;
    zero_f4<<<(n4 + 255) / 256, 256, 0, stream>>>((float4*)out, n4);

    dim3 grid(TILES, NUM_K);
    kan_subm_conv3d<<<grid, 256, 0, stream>>>(features, base_w, spline_w,
                                              inp_idx, out_idx, out);
}
